// LSTMCell_90958817394974
// MI455X (gfx1250) — matrix operand order repaired and checked by reading, not compiled, not run on hardware
//
#include <hip/hip_runtime.h>

typedef __attribute__((ext_vector_type(16))) __bf16 v16bf;
typedef __attribute__((ext_vector_type(8)))  float  v8f;
typedef __attribute__((ext_vector_type(4)))  unsigned int u32x4;

#define BDIM  256
#define BM    256          // block rows (batch); 8 waves x 32 rows
#define BN    32           // block cols (hidden)
#define KB    32           // K per stage
#define KDIM  1024
#define NSTEP 64           // (I + H) / KB = 2048/32

// Pack two f32 -> packed bf16 pair (round-half-up): 2x v_add + 1x v_perm_b32.
__device__ __forceinline__ unsigned pk2(float lo, float hi) {
    unsigned ul = __builtin_bit_cast(unsigned, lo) + 0x8000u;
    unsigned uh = __builtin_bit_cast(unsigned, hi) + 0x8000u;
#if __has_builtin(__builtin_amdgcn_perm)
    return __builtin_amdgcn_perm(uh, ul, 0x07060302u);  // {uh[31:16], ul[31:16]}
#else
    return (uh & 0xFFFF0000u) | (ul >> 16);
#endif
}
__device__ __forceinline__ v16bf mkfrag(u32x4 lo, u32x4 hi) {
    union { u32x4 u[2]; v16bf v; } x; x.u[0] = lo; x.u[1] = hi; return x.v;
}
__device__ __forceinline__ float sigm(float x)      { return 1.0f / (1.0f + __expf(-x)); }
__device__ __forceinline__ float tanh_fast(float x) { return 1.0f - 2.0f / (__expf(2.0f * x) + 1.0f); }

__device__ __forceinline__ v8f wmma_bf16(v16bf a, v16bf b, v8f c) {
    return __builtin_amdgcn_wmma_f32_16x16x32_bf16(false, a, false, b, (short)0, c,
                                                   false, false);
}

__global__ __launch_bounds__(BDIM)
void lstm_cell_wmma_bf16(const float* __restrict__ X,   // [B, I] incoming
                         const float* __restrict__ Hs,  // [B, H] h
                         const float* __restrict__ Cs,  // [B, H] c
                         const float* __restrict__ Wx,  // [4, H, I]
                         const float* __restrict__ Wh,  // [4, H, H]
                         const float* __restrict__ bx,  // [4, H]
                         const float* __restrict__ bh,  // [4, H]
                         float* __restrict__ out)       // [3, B, H] = (h, h, c)
{
    __shared__ __align__(16) unsigned short sA[2][BM * KB];      // 2 x 16KB bf16
    __shared__ __align__(16) unsigned short sW[2][4 * BN * KB];  // 2 x  8KB bf16

    const int tid  = threadIdx.x;
    const int wave = tid >> 5;
    const int lane = tid & 31;
    const int half = lane >> 4;    // K-half selector for A/B fragments
    const int ln   = lane & 15;    // row (A:M, B:N) within 16
    const long rbase = (long)blockIdx.x * BM;   // batch-row base
    const int  hbase = blockIdx.y * BN;         // hidden-col base

    // ---- global-load staging registers (double-buffer pipeline) ----
    float4 rA[8], rW[4];
    const int ldr = tid >> 3;      // 0..31
    const int ldc = (tid & 7) * 4; // 0,4,...,28

    auto gload = [&](int step) {
        const float* Ap = (step < 32) ? X  : Hs;
        const float* Wp = (step < 32) ? Wx : Wh;
        const int k0 = (step & 31) * KB;
#pragma unroll
        for (int p = 0; p < 8; ++p) {
            const long r = rbase + p * 32 + ldr;
            rA[p] = *(const float4*)(Ap + r * KDIM + k0 + ldc);
        }
#pragma unroll
        for (int p = 0; p < 4; ++p) {  // p == gate index
            const long hr = hbase + ldr;
            rW[p] = *(const float4*)(Wp + (long)p * KDIM * KDIM + hr * KDIM + k0 + ldc);
        }
    };

    auto stor = [&](int buf) {
#pragma unroll
        for (int p = 0; p < 8; ++p) {
            uint2 v; v.x = pk2(rA[p].x, rA[p].y); v.y = pk2(rA[p].z, rA[p].w);
            *(uint2*)&sA[buf][(p * 32 + ldr) * KB + ldc] = v;
        }
#pragma unroll
        for (int p = 0; p < 4; ++p) {
            uint2 v; v.x = pk2(rW[p].x, rW[p].y); v.y = pk2(rW[p].z, rW[p].w);
            *(uint2*)&sW[buf][p * BN * KB + ldr * KB + ldc] = v;
        }
    };

    const v8f vzero = {0.f, 0.f, 0.f, 0.f, 0.f, 0.f, 0.f, 0.f};
    v8f acc0[8], acc1[8];               // [gate*2 + ntile], rows 0-15 / 16-31
#pragma unroll
    for (int b = 0; b < 8; ++b) { acc0[b] = vzero; acc1[b] = vzero; }

    gload(0);
#pragma unroll 1
    for (int s = 0; s < NSTEP; ++s) {
        const int buf = s & 1;
        stor(buf);
        // Single barrier per iteration: separates this stage's LDS stores from
        // fragment reads, AND (two iterations apart) reads of a buffer from its
        // overwrite. A second loop-end barrier would force loadcnt==0 on the
        // in-flight prefetch every iteration -- deliberately omitted.
        __syncthreads();
        if (s + 1 < NSTEP) gload(s + 1); // in-flight global loads overlap WMMAs

        // A fragments: two 16x32 bf16 slices (wave rows 0-15, 16-31)
        const int ar0 = (wave * 32 + ln) * KB;
        const int ar1 = (wave * 32 + 16 + ln) * KB;
        const v16bf a0 = mkfrag(*(const u32x4*)&sA[buf][ar0 + half * 8],
                                *(const u32x4*)&sA[buf][ar0 + half * 8 + 16]);
        const v16bf a1 = mkfrag(*(const u32x4*)&sA[buf][ar1 + half * 8],
                                *(const u32x4*)&sA[buf][ar1 + half * 8 + 16]);
#pragma unroll
        for (int b = 0; b < 8; ++b) {    // b = gate*2 + ntile
            const int g = b >> 1, j = b & 1;
            const int brow = g * BN * KB + (j * 16 + ln) * KB + half * 8;
            const v16bf bf = mkfrag(*(const u32x4*)&sW[buf][brow],
                                    *(const u32x4*)&sW[buf][brow + 16]);
            acc0[b] = wmma_bf16(a0, bf, acc0[b]);
            acc1[b] = wmma_bf16(a1, bf, acc1[b]);
        }

#if __has_builtin(__builtin_amdgcn_sched_group_barrier)
        // Pipeline prescription for this scheduling region:
        //   12 DS reads (A frags + 4 B frags) -> 12 global prefetch loads ->
        //   8 WMMAs -> remaining 8 DS reads -> 8 WMMAs
        __builtin_amdgcn_sched_group_barrier(0x100, 12, 0); // DS read
        __builtin_amdgcn_sched_group_barrier(0x020, 12, 0); // VMEM read
        __builtin_amdgcn_sched_group_barrier(0x008,  8, 0); // WMMA
        __builtin_amdgcn_sched_group_barrier(0x100,  8, 0); // DS read
        __builtin_amdgcn_sched_group_barrier(0x008,  8, 0); // WMMA
#endif
    }

    // ---- fused LSTM epilogue: gates -> (new_h, new_h, new_c) ----
    const long BH = (long)8192 * 1024;
#pragma unroll
    for (int j = 0; j < 2; ++j) {
        const int hc = hbase + j * 16 + ln;
        const float bi = bx[0 * KDIM + hc] + bh[0 * KDIM + hc];
        const float bf = bx[1 * KDIM + hc] + bh[1 * KDIM + hc];
        const float bg = bx[2 * KDIM + hc] + bh[2 * KDIM + hc];
        const float bo = bx[3 * KDIM + hc] + bh[3 * KDIM + hc];
#pragma unroll
        for (int t = 0; t < 2; ++t) {
            const v8f* ac = t ? acc1 : acc0;
#pragma unroll
            for (int r = 0; r < 8; ++r) {
                const long row = rbase + wave * 32 + t * 16 + half * 8 + r;
                const long off = row * KDIM + hc;
                const float ig = sigm(ac[0 * 2 + j][r] + bi);
                const float fg = sigm(ac[1 * 2 + j][r] + bf);
                const float gg = tanh_fast(ac[2 * 2 + j][r] + bg);
                const float og = sigm(ac[3 * 2 + j][r] + bo);
                const float cn = fg * Cs[off] + ig * gg;
                const float hn = og * tanh_fast(cn);
                out[off]          = hn;   // new_h  (output #1)
                out[BH + off]     = hn;   // new_h  (output #2)
                out[2 * BH + off] = cn;   // new_c  (output #3)
            }
        }
    }
}

extern "C" void kernel_launch(void* const* d_in, const int* in_sizes, int n_in,
                              void* d_out, int out_size, void* d_ws, size_t ws_size,
                              hipStream_t stream) {
    const float* X  = (const float*)d_in[0];  // incoming [8192,1024]
    const float* Hs = (const float*)d_in[1];  // h        [8192,1024]
    const float* Cs = (const float*)d_in[2];  // c        [8192,1024]
    const float* Wx = (const float*)d_in[3];  // [4,1024,1024]
    const float* Wh = (const float*)d_in[4];  // [4,1024,1024]
    const float* bx = (const float*)d_in[5];  // [4,1024]
    const float* bh = (const float*)d_in[6];  // [4,1024]
    float* out = (float*)d_out;               // [3, 8192, 1024]

    dim3 grid(8192 / BM, 1024 / BN);          // 32 x 32 blocks
    lstm_cell_wmma_bf16<<<grid, BDIM, 0, stream>>>(X, Hs, Cs, Wx, Wh, bx, bh, out);
}
